// ContrastiveLoss_6571299963520
// MI455X (gfx1250) — matrix operand order repaired and checked by reading, not compiled, not run on hardware
//
#include <hip/hip_runtime.h>
#include <hip/hip_bf16.h>

typedef _Float16 v8h  __attribute__((ext_vector_type(8)));
typedef _Float16 v16h __attribute__((ext_vector_type(16)));
typedef float    v8f  __attribute__((ext_vector_type(8)));

#define BATCH 2048
#define DIM   768
#define N2    4096        // 2*BATCH
#define TEMP_INV 2.0f     // 1/TEMP, TEMP = 0.5
#define BAL   0.01f

// ---------------------------------------------------------------------------
// Kernel 1: L2-normalize each row of [emb_i; emb_j] and store as f16 reps.
// One block per row (4096 blocks, 256 threads; 768 = 3 * 256).
// Block 0 / thread 0 also zeroes the scalar output accumulator.
// ---------------------------------------------------------------------------
__global__ __launch_bounds__(256) void norm_kernel(
    const float* __restrict__ emb_i, const float* __restrict__ emb_j,
    _Float16* __restrict__ reps, float* __restrict__ out)
{
    const int r   = blockIdx.x;
    const int tid = threadIdx.x;
    const float* src = (r < BATCH) ? (emb_i + (size_t)r * DIM)
                                   : (emb_j + (size_t)(r - BATCH) * DIM);
    const float x0 = src[tid];
    const float x1 = src[tid + 256];
    const float x2 = src[tid + 512];
    float ss = x0 * x0 + x1 * x1 + x2 * x2;

    // wave32 butterfly reduce, then cross-wave via LDS
    #pragma unroll
    for (int m = 16; m >= 1; m >>= 1) ss += __shfl_xor(ss, m, 32);
    __shared__ float red[8];
    if ((tid & 31) == 0) red[tid >> 5] = ss;
    __syncthreads();
    float tot = 0.0f;
    #pragma unroll
    for (int w = 0; w < 8; ++w) tot += red[w];

    const float scale = 1.0f / fmaxf(sqrtf(tot), 1e-12f);
    _Float16* dst = reps + (size_t)r * DIM;
    dst[tid]       = (_Float16)(x0 * scale);
    dst[tid + 256] = (_Float16)(x1 * scale);
    dst[tid + 512] = (_Float16)(x2 * scale);

    if (r == 0 && tid == 0) out[0] = 0.0f;
}

// ---------------------------------------------------------------------------
// Kernel 2: fused sim-GEMM + contrastive loss reductions.
// 256 blocks (one 16-row tile each), 256 threads = 8 wave32 waves.
// A tile (16 x 768 f16 = 24 KB) lives in LDS; each wave streams 32 of the
// 256 column tiles, 24 x v_wmma_f32_16x16x32_f16 per 16x16 tile (K = 768).
// Per-row statistics accumulated in LDS, folded into d_out via global atomic.
// ---------------------------------------------------------------------------
__global__ __launch_bounds__(256) void simloss_kernel(
    const _Float16* __restrict__ reps, const int* __restrict__ pcls,
    float* __restrict__ out)
{
    __shared__ __align__(16) _Float16 As[16 * DIM];  // 24 KB A tile
    __shared__ float acc[4][16];  // [0]=offdiag exp-sum, [1]=den_v, [2]=same-sim, [3]=pos
    __shared__ int   rcls[16];

    const int tid     = threadIdx.x;
    const int rowBase = blockIdx.x * 16;

    if (tid < 64) acc[tid >> 4][tid & 15] = 0.0f;
    if (tid < 16) rcls[tid] = pcls[(rowBase + tid) & (BATCH - 1)];

    {   // cooperative A-tile load: 16*768/8 = 1536 16-byte chunks
        const v8h* g = (const v8h*)(reps + (size_t)rowBase * DIM);
        v8h* s = (v8h*)As;
        #pragma unroll
        for (int i = tid; i < 16 * DIM / 8; i += 256) s[i] = g[i];
    }
    __syncthreads();

    const int wave = tid >> 5;
    const int lane = tid & 31;
    const int half = lane >> 4;   // 0: lanes 0-15, 1: lanes 16-31
    const int l16  = lane & 15;

    for (int ct = wave; ct < N2 / 16; ct += 8) {
        const int colBase = ct * 16;
        // B fragment source: row (colBase + l16) of reps; K chunk per half
        const _Float16* brow = reps + (size_t)(colBase + l16) * DIM + half * 8;
        // A fragment source: LDS row l16; K = kb + half*8 + {0..7} and +16
        const _Float16* arow = As + l16 * DIM + half * 8;

        v8f c = {};
        #pragma unroll
        for (int kb = 0; kb < DIM; kb += 32) {
            v16h a, b;
            *((v8h*)&a)     = *(const v8h*)(arow + kb);        // K: kb+h8+0..7
            *((v8h*)&a + 1) = *(const v8h*)(arow + kb + 16);   // K: kb+h8+16..23
            *((v8h*)&b)     = *(const v8h*)(brow + kb);        // K: kb+h8+0..7
            *((v8h*)&b + 1) = *(const v8h*)(brow + kb + 16);   // K: kb+h8+16..23
            c = __builtin_amdgcn_wmma_f32_16x16x32_f16(
                    false, a, false, b, (short)0, c, false, false);
        }

        // Epilogue: C layout -> VGPR v holds M = v + half*8, lane%16 = N.
        const int cglob = colBase + l16;
        const int ccls  = pcls[cglob & (BATCH - 1)];
        #pragma unroll
        for (int v = 0; v < 8; ++v) {
            const int  rglob = rowBase + half * 8 + v;
            const float s    = c[v];
            const bool same  = (ccls == rcls[half * 8 + v]);
            const bool diag  = (cglob == rglob);
            const float e    = __expf(TEMP_INV * s);
            float t1 = diag ? 0.0f : e;                 // NT-Xent denominator
            float t2 = same ? 1.0f : e;                 // class-loss denominator
            float t3 = (same && !diag) ? s : 0.0f;      // same-class sim sum
            float t4 = (cglob == ((rglob + BATCH) & (N2 - 1))) ? s : 0.0f; // pos pair
            #pragma unroll
            for (int m = 1; m < 16; m <<= 1) {          // reduce over N (16 lanes)
                t1 += __shfl_xor(t1, m, 32);
                t2 += __shfl_xor(t2, m, 32);
                t3 += __shfl_xor(t3, m, 32);
                t4 += __shfl_xor(t4, m, 32);
            }
            if (l16 == 0) {                             // lanes 0 and 16
                atomicAdd(&acc[0][half * 8 + v], t1);
                atomicAdd(&acc[1][half * 8 + v], t2);
                atomicAdd(&acc[2][half * 8 + v], t3);
                atomicAdd(&acc[3][half * 8 + v], t4);
            }
        }
    }
    __syncthreads();

    if (tid < 16) {
        const float S1 = acc[0][tid];  // sum_{j!=i} exp(sim/T)
        const float S2 = acc[1][tid];  // den_v_i
        const float S3 = acc[2][tid];  // sum same-class off-diag sim
        const float P  = acc[3][tid];  // positive-pair sim
        float term = (__logf(S1) - TEMP_INV * P)
                   + BAL * ((float)N2 * __logf(S2) - TEMP_INV * S3);
        term *= (1.0f / (2.0f * (float)N2));            // /8192
        #pragma unroll
        for (int m = 1; m < 16; m <<= 1) term += __shfl_xor(term, m, 32);
        if (tid == 0) atomicAdd(out, term);
    }
}

extern "C" void kernel_launch(void* const* d_in, const int* in_sizes, int n_in,
                              void* d_out, int out_size, void* d_ws, size_t ws_size,
                              hipStream_t stream) {
    const float* emb_i = (const float*)d_in[0];
    const float* emb_j = (const float*)d_in[1];
    const int*   pcls  = (const int*)d_in[2];
    float*       out   = (float*)d_out;
    _Float16*    reps  = (_Float16*)d_ws;   // 4096*768*2 = 6 MB scratch

    norm_kernel<<<N2, 256, 0, stream>>>(emb_i, emb_j, reps, out);
    simloss_kernel<<<N2 / 16, 256, 0, stream>>>(reps, pcls, out);
}
